// TransformerBlock_249108103260
// MI455X (gfx1250) — matrix operand order repaired and checked by reading, not compiled, not run on hardware
//
#include <hip/hip_runtime.h>
#include <hip/hip_bf16.h>

#define Bb 4
#define Ss 2048
#define Dd 768
#define Hh 12
#define DHd 64
#define FFf 3072
#define BSr (Bb*Ss)

typedef __bf16 bf16_t;
typedef __attribute__((ext_vector_type(16))) __bf16 v16bf;
typedef __attribute__((ext_vector_type(8)))  float  v8f;

union AFrag { v16bf v; uint4 q[2]; };

__device__ __forceinline__ bf16_t f2bf(float f) {
  unsigned u = __builtin_bit_cast(unsigned, f);
  unsigned r = u + 0x7fffu + ((u >> 16) & 1u);
  unsigned short h = (unsigned short)(r >> 16);
  return __builtin_bit_cast(bf16_t, h);
}

// ---- CDNA5 async global->LDS copy (16B per lane), guarded fallback ----
#if __has_builtin(__builtin_amdgcn_global_load_async_to_lds_b128)
#define HAVE_ASYNC_LDS 1
typedef int v4i_vs __attribute__((__vector_size__(16)));
typedef __attribute__((address_space(1))) v4i_vs* as1_v4i_p;
typedef __attribute__((address_space(3))) v4i_vs* as3_v4i_p;
__device__ __forceinline__ void async_copy16(const void* g, void* l) {
  __builtin_amdgcn_global_load_async_to_lds_b128(
      (as1_v4i_p)(unsigned long long)g,
      (as3_v4i_p)(unsigned long long)l,
      0, 0);
}
__device__ __forceinline__ void async_wait0() {
#if __has_builtin(__builtin_amdgcn_s_wait_asynccnt)
  __builtin_amdgcn_s_wait_asynccnt(0);
#else
  asm volatile("s_wait_asynccnt 0" ::: "memory");
#endif
}
#else
__device__ __forceinline__ void async_copy16(const void* g, void* l) {
  *(uint4*)l = *(const uint4*)g;
}
__device__ __forceinline__ void async_wait0() {}
#endif

// ---------------- LayerNorm (one row per block) + bf16 cast ----------------
__global__ __launch_bounds__(256)
void k_ln(const float* __restrict__ x, const float* __restrict__ g,
          const float* __restrict__ bb, bf16_t* __restrict__ out) {
  int row = blockIdx.x;
  int t = threadIdx.x;
  const float* xr = x + (size_t)row * Dd;
  float v0 = xr[t], v1 = xr[t + 256], v2 = xr[t + 512];
  __shared__ float r1[256], r2[256];
  r1[t] = v0 + v1 + v2;
  r2[t] = v0 * v0 + v1 * v1 + v2 * v2;
  __syncthreads();
  for (int off = 128; off > 0; off >>= 1) {
    if (t < off) { r1[t] += r1[t + off]; r2[t] += r2[t + off]; }
    __syncthreads();
  }
  float mean = r1[0] * (1.0f / Dd);
  float var  = r2[0] * (1.0f / Dd) - mean * mean;
  float rstd = rsqrtf(var + 1e-5f);
  bf16_t* orow = out + (size_t)row * Dd;
  orow[t]       = f2bf((v0 - mean) * rstd * g[t]       + bb[t]);
  orow[t + 256] = f2bf((v1 - mean) * rstd * g[t + 256] + bb[t + 256]);
  orow[t + 512] = f2bf((v2 - mean) * rstd * g[t + 512] + bb[t + 512]);
}

// ---------------- weight casts ----------------
__global__ void k_cast(const float* __restrict__ in, bf16_t* __restrict__ out, int n) {
  int i = blockIdx.x * 256 + threadIdx.x;
  if (i < n) out[i] = f2bf(in[i]);
}
// [H,D,DH] -> row-major [D, H*DH]
__global__ void k_cast_qkv(const float* __restrict__ in, bf16_t* __restrict__ out) {
  int i = blockIdx.x * 256 + threadIdx.x;      // i = d*768 + j
  if (i >= Dd * Dd) return;
  int d = i / Dd, j = i % Dd;
  int h = j >> 6, e = j & 63;
  out[i] = f2bf(in[((size_t)h * Dd + d) * DHd + e]);
}

// ---------------- generic bf16 WMMA GEMM ----------------
// C[M,N] = A[M,K]*B[K,N] + bias (+GeLU) (+residual) -> f32 / bf16 stores.
// Block tile 128x64, 8 waves, wave tile 32x32, K step 32.
// Double-buffered LDS; A tile via async global->LDS, B tile via register
// pipeline + transposed LDS stores overlapped with WMMA.
template <int ACT, bool RES, bool OUTF, bool OUTB>
__global__ __launch_bounds__(256)
void k_gemm(const bf16_t* __restrict__ A, const bf16_t* __restrict__ Bm,
            const float* __restrict__ bias, const float* __restrict__ resid,
            float* __restrict__ outF, bf16_t* __restrict__ outB,
            int M, int N, int K) {
  __shared__ bf16_t sA[2][128 * 32];   // [m][k] row-major
  __shared__ bf16_t sB[2][64 * 32];    // [n][k] transposed
  int tid = threadIdx.x;
  int lane = tid & 31, w = tid >> 5;
  int wm = w >> 1, wn = w & 1;
  int m0 = blockIdx.y * 128, n0 = blockIdx.x * 64;
  v8f acc[2][2] = {};
  int kb = (lane >> 4) * 8;     // A fragment K-chunk base
  int ks = (lane >> 4) * 16;    // unused; B fragment also uses kb

  // this thread's copy slots
  int arow[2], acol[2];
#pragma unroll
  for (int i = 0; i < 2; ++i) { int c = tid + i * 256; arow[i] = c >> 2; acol[i] = (c & 3) * 8; }
  int brow[8], bcol[8];
#pragma unroll
  for (int i = 0; i < 8; ++i) { int l = tid + i * 256; brow[i] = l >> 6; bcol[i] = l & 63; }

  // ---- prologue: stage tile 0 ----
#pragma unroll
  for (int i = 0; i < 2; ++i)
    async_copy16(&A[(size_t)(m0 + arow[i]) * K + acol[i]], &sA[0][arow[i] * 32 + acol[i]]);
#pragma unroll
  for (int i = 0; i < 8; ++i)
    sB[0][bcol[i] * 32 + brow[i]] = Bm[(size_t)brow[i] * N + n0 + bcol[i]];
  async_wait0();
  __syncthreads();

  int buf = 0;
  for (int kk = 0; kk < K; kk += 32) {
    int nxt = kk + 32;
    bool has = nxt < K;
    bf16_t breg[8];
    if (has) {
      // issue next B global loads early (latency hidden behind WMMA)
#pragma unroll
      for (int i = 0; i < 8; ++i)
        breg[i] = Bm[(size_t)(nxt + brow[i]) * N + n0 + bcol[i]];
      // async next A tile into the other buffer
#pragma unroll
      for (int i = 0; i < 2; ++i)
        async_copy16(&A[(size_t)(m0 + arow[i]) * K + nxt + acol[i]],
                     &sA[buf ^ 1][arow[i] * 32 + acol[i]]);
      if (nxt + 32 < K)
        __builtin_prefetch(&Bm[(size_t)(nxt + 32 + brow[0]) * N + n0 + bcol[0]], 0, 1);
    }

    AFrag a[2], b[2];
#pragma unroll
    for (int mi = 0; mi < 2; ++mi) {
      const bf16_t* pr = &sA[buf][(wm * 32 + mi * 16 + (lane & 15)) * 32];
      a[mi].q[0] = *(const uint4*)(pr + kb);
      a[mi].q[1] = *(const uint4*)(pr + kb + 16);
    }
#pragma unroll
    for (int ni = 0; ni < 2; ++ni) {
      const bf16_t* pr = &sB[buf][(wn * 32 + ni * 16 + (lane & 15)) * 32];
      b[ni].q[0] = *(const uint4*)(pr + kb);
      b[ni].q[1] = *(const uint4*)(pr + kb + 16);
    }
#pragma unroll
    for (int mi = 0; mi < 2; ++mi)
#pragma unroll
      for (int ni = 0; ni < 2; ++ni)
        acc[mi][ni] = __builtin_amdgcn_wmma_f32_16x16x32_bf16(
            false, a[mi].v, false, b[ni].v, (short)0, acc[mi][ni], false, false);

    if (has) {
#pragma unroll
      for (int i = 0; i < 8; ++i)
        sB[buf ^ 1][bcol[i] * 32 + brow[i]] = breg[i];
    }
    async_wait0();
    __syncthreads();
    buf ^= 1;
  }

  int rofs = (lane >> 4) * 8;
#pragma unroll
  for (int mi = 0; mi < 2; ++mi)
#pragma unroll
    for (int ni = 0; ni < 2; ++ni) {
      int col = n0 + wn * 32 + ni * 16 + (lane & 15);
      float bv = bias[col];
#pragma unroll
      for (int r = 0; r < 8; ++r) {
        int row = m0 + wm * 32 + mi * 16 + rofs + r;
        float v = acc[mi][ni][r] + bv;
        if (ACT) v = 0.5f * v * (1.0f + erff(v * 0.70710678118f));
        if (RES) v += resid[(size_t)row * N + col];
        if (OUTF) outF[(size_t)row * N + col] = v;
        if (OUTB) outB[(size_t)row * N + col] = f2bf(v);
      }
    }
}

// ---------------- fused flash attention ----------------
// grid (S/64, H, B), 128 threads = 4 waves; wave owns 16 q rows; kv chunk 32.
__global__ __launch_bounds__(128)
void k_attn(const bf16_t* __restrict__ q, const bf16_t* __restrict__ kM,
            const bf16_t* __restrict__ vM, bf16_t* __restrict__ o) {
  __shared__ bf16_t sV[64 * 32];        // transposed V chunk [e][kv]
  __shared__ bf16_t sP[4][16 * 32];     // per-wave probs [row][kv]
  int tid = threadIdx.x;
  int lane = tid & 31, w = tid >> 5;
  int b = blockIdx.z, h = blockIdx.y, qt = blockIdx.x;
  int kb = (lane >> 4) * 8;
  int ks = (lane >> 4) * 16;

  int qrow = qt * 64 + w * 16 + (lane & 15);
  const bf16_t* qp = q + ((size_t)(b * Ss + qrow)) * Dd + h * DHd;
  AFrag a0, a1;
  a0.q[0] = *(const uint4*)(qp + kb);       a0.q[1] = *(const uint4*)(qp + kb + 16);
  a1.q[0] = *(const uint4*)(qp + 32 + kb);  a1.q[1] = *(const uint4*)(qp + 32 + kb + 16);

  float mrun[8], lrun[8];
#pragma unroll
  for (int r = 0; r < 8; ++r) { mrun[r] = -3.0e38f; lrun[r] = 0.0f; }
  v8f oacc[4] = {};
  const float scale = 0.03608439182435161f;  // 1/sqrt(768) per reference

  for (int kt = 0; kt < Ss / 32; ++kt) {
#pragma unroll
    for (int i = 0; i < 16; ++i) {           // stage V chunk transposed
      int l = tid + i * 128;
      int kv = l >> 6, e = l & 63;
      sV[e * 32 + kv] = vM[((size_t)(b * Ss + kt * 32 + kv)) * Dd + h * DHd + e];
    }
    __syncthreads();

    float sc[2][8];
#pragma unroll
    for (int st = 0; st < 2; ++st) {         // scores, 16 kv cols at a time
      int kvr = kt * 32 + st * 16 + (lane & 15);
      const bf16_t* kp = kM + ((size_t)(b * Ss + kvr)) * Dd + h * DHd;
      AFrag b0, b1;
      b0.q[0] = *(const uint4*)(kp + kb);       b0.q[1] = *(const uint4*)(kp + kb + 16);
      b1.q[0] = *(const uint4*)(kp + 32 + kb);  b1.q[1] = *(const uint4*)(kp + 32 + kb + 16);
      v8f c = {};
      c = __builtin_amdgcn_wmma_f32_16x16x32_bf16(false, a0.v, false, b0.v, (short)0, c, false, false);
      c = __builtin_amdgcn_wmma_f32_16x16x32_bf16(false, a1.v, false, b1.v, (short)0, c, false, false);
#pragma unroll
      for (int r = 0; r < 8; ++r) sc[st][r] = c[r] * scale;
    }

    bf16_t* pw = &sP[w][0];
#pragma unroll
    for (int r = 0; r < 8; ++r) {            // online softmax per row
      float mloc = fmaxf(sc[0][r], sc[1][r]);
#pragma unroll
      for (int d = 8; d >= 1; d >>= 1) mloc = fmaxf(mloc, __shfl_xor(mloc, d, 32));
      float mnew = fmaxf(mrun[r], mloc);
      float corr = __expf(mrun[r] - mnew);
      float p0 = __expf(sc[0][r] - mnew);
      float p1 = __expf(sc[1][r] - mnew);
      float lsum = p0 + p1;
#pragma unroll
      for (int d = 8; d >= 1; d >>= 1) lsum += __shfl_xor(lsum, d, 32);
      lrun[r] = lrun[r] * corr + lsum;
      mrun[r] = mnew;
#pragma unroll
      for (int ni = 0; ni < 4; ++ni) oacc[ni][r] *= corr;
      int prow = r + (lane >> 4) * 8;
      pw[prow * 32 + (lane & 15)]      = f2bf(p0);
      pw[prow * 32 + 16 + (lane & 15)] = f2bf(p1);
    }

    AFrag ap;                                 // P as A-fragment (same wave, LDS in-order)
    {
      const bf16_t* pr = &pw[(lane & 15) * 32];
      ap.q[0] = *(const uint4*)(pr + kb);
      ap.q[1] = *(const uint4*)(pr + kb + 16);
    }
#pragma unroll
    for (int ni = 0; ni < 4; ++ni) {          // o += P @ V
      const bf16_t* pr = &sV[(ni * 16 + (lane & 15)) * 32];
      AFrag bv;
      bv.q[0] = *(const uint4*)(pr + kb);
      bv.q[1] = *(const uint4*)(pr + kb + 16);
      oacc[ni] = __builtin_amdgcn_wmma_f32_16x16x32_bf16(
          false, ap.v, false, bv.v, (short)0, oacc[ni], false, false);
    }
    __syncthreads();
  }

  int rofs = (lane >> 4) * 8;
#pragma unroll
  for (int ni = 0; ni < 4; ++ni) {
    int col = h * DHd + ni * 16 + (lane & 15);
#pragma unroll
    for (int r = 0; r < 8; ++r) {
      int row = b * Ss + qt * 64 + w * 16 + rofs + r;
      o[(size_t)row * Dd + col] = f2bf(oacc[ni][r] / lrun[r]);
    }
  }
}

// ---------------- host launch ----------------
extern "C" void kernel_launch(void* const* d_in, const int* in_sizes, int n_in,
                              void* d_out, int out_size, void* d_ws, size_t ws_size,
                              hipStream_t stream) {
  const float* x     = (const float*)d_in[0];
  const float* ln1_g = (const float*)d_in[1];
  const float* ln1_b = (const float*)d_in[2];
  const float* Wq    = (const float*)d_in[3];
  const float* bq    = (const float*)d_in[4];
  const float* Wk    = (const float*)d_in[5];
  const float* bk    = (const float*)d_in[6];
  const float* Wv    = (const float*)d_in[7];
  const float* bv    = (const float*)d_in[8];
  const float* Wo    = (const float*)d_in[9];
  const float* bo    = (const float*)d_in[10];
  const float* ln2_g = (const float*)d_in[11];
  const float* ln2_b = (const float*)d_in[12];
  const float* W1    = (const float*)d_in[13];
  const float* b1    = (const float*)d_in[14];
  const float* W2    = (const float*)d_in[15];
  const float* b2    = (const float*)d_in[16];

  size_t off = 0;
  char* base = (char*)d_ws;
  auto alloc = [&](size_t bytes) -> char* {
    char* p = base + off;
    off = (off + bytes + 255) & ~(size_t)255;
    return p;
  };
  bf16_t* Wq2 = (bf16_t*)alloc((size_t)Dd * Dd * 2);
  bf16_t* Wk2 = (bf16_t*)alloc((size_t)Dd * Dd * 2);
  bf16_t* Wv2 = (bf16_t*)alloc((size_t)Dd * Dd * 2);
  bf16_t* Wo2 = (bf16_t*)alloc((size_t)Dd * Dd * 2);
  bf16_t* W1b = (bf16_t*)alloc((size_t)Dd * FFf * 2);
  bf16_t* W2b = (bf16_t*)alloc((size_t)FFf * Dd * 2);
  bf16_t* hB  = (bf16_t*)alloc((size_t)BSr * Dd * 2);
  float*  x1  = (float*) alloc((size_t)BSr * Dd * 4);
  bf16_t* qB  = (bf16_t*)alloc((size_t)BSr * Dd * 2);
  bf16_t* kB  = (bf16_t*)alloc((size_t)BSr * Dd * 2);
  bf16_t* vB  = (bf16_t*)alloc((size_t)BSr * Dd * 2);
  bf16_t* oB  = (bf16_t*)alloc((size_t)BSr * Dd * 2);
  bf16_t* m1B = qB;   // alias: q/k/v/o dead after Wo GEMM; 4*12.58MB == 50.3MB

  int nDD = Dd * Dd, nDF = Dd * FFf;
  // weights -> bf16
  k_cast_qkv<<<(nDD + 255) / 256, 256, 0, stream>>>(Wq, Wq2);
  k_cast_qkv<<<(nDD + 255) / 256, 256, 0, stream>>>(Wk, Wk2);
  k_cast_qkv<<<(nDD + 255) / 256, 256, 0, stream>>>(Wv, Wv2);
  k_cast<<<(nDD + 255) / 256, 256, 0, stream>>>(Wo, Wo2, nDD);
  k_cast<<<(nDF + 255) / 256, 256, 0, stream>>>(W1, W1b, nDF);
  k_cast<<<(nDF + 255) / 256, 256, 0, stream>>>(W2, W2b, nDF);
  // LN1
  k_ln<<<BSr, 256, 0, stream>>>(x, ln1_g, ln1_b, hB);
  // QKV projections
  dim3 gQ(Dd / 64, BSr / 128);
  k_gemm<0, false, false, true><<<gQ, 256, 0, stream>>>(
      hB, Wq2, bq, nullptr, nullptr, qB, BSr, Dd, Dd);
  k_gemm<0, false, false, true><<<gQ, 256, 0, stream>>>(
      hB, Wk2, bk, nullptr, nullptr, kB, BSr, Dd, Dd);
  k_gemm<0, false, false, true><<<gQ, 256, 0, stream>>>(
      hB, Wv2, bv, nullptr, nullptr, vB, BSr, Dd, Dd);
  // fused attention
  k_attn<<<dim3(Ss / 64, Hh, Bb), 128, 0, stream>>>(qB, kB, vB, oB);
  // output projection + residual -> x1 (f32)
  k_gemm<0, true, true, false><<<gQ, 256, 0, stream>>>(
      oB, Wo2, bo, x, x1, nullptr, BSr, Dd, Dd);
  // LN2 (reuse hB)
  k_ln<<<BSr, 256, 0, stream>>>(x1, ln2_g, ln2_b, hB);
  // MLP
  k_gemm<1, false, false, true><<<dim3(FFf / 64, BSr / 128), 256, 0, stream>>>(
      hB, W1b, b1, nullptr, nullptr, m1B, BSr, FFf, Dd);
  k_gemm<0, true, true, false><<<gQ, 256, 0, stream>>>(
      m1B, W2b, b2, x1, (float*)d_out, nullptr, BSr, Dd, FFf);
}
